// CNN_MoE_v1_73143293051399
// MI455X (gfx1250) — matrix operand order repaired and checked by reading, not compiled, not run on hardware
//
#include <hip/hip_runtime.h>

// ---------------- Types / helpers ----------------
typedef __attribute__((ext_vector_type(16))) __bf16 v16bf;
typedef __attribute__((ext_vector_type(8)))  float  v8f;

union FragCvt { uint4 u[2]; v16bf v; };

__device__ __forceinline__ v16bf make_frag(uint4 lo, uint4 hi) {
  FragCvt f; f.u[0] = lo; f.u[1] = hi; return f.v;
}

// f32 -> bf16, round-to-nearest-even
__device__ __forceinline__ unsigned short f2bf(float f) {
  unsigned int u = __float_as_uint(f);
  u += 0x7FFFu + ((u >> 16) & 1u);
  return (unsigned short)(u >> 16);
}

// ---------------- TDM (Tensor Data Mover) support ----------------
#if defined(__has_builtin)
#if __has_builtin(__builtin_amdgcn_tensor_load_to_lds) && __has_builtin(__builtin_amdgcn_s_wait_tensorcnt)
#define CDNA5_HAS_TDM 1
#endif
#endif
#ifndef CDNA5_HAS_TDM
#define CDNA5_HAS_TDM 0
#endif

#if CDNA5_HAS_TDM
typedef __attribute__((ext_vector_type(4))) unsigned int u32x4;
typedef __attribute__((ext_vector_type(8))) int          i32x8;
typedef __attribute__((ext_vector_type(4))) int          i32x4;

// 2-D tile load, 2-byte elements, row-major into LDS.
// D# packing per CDNA5 ISA.
__device__ __forceinline__ void tdm_load_2d(unsigned lds_addr, unsigned long long gaddr,
                                            unsigned td0, unsigned td1,
                                            unsigned tile0, unsigned tile1,
                                            unsigned stride0) {
  u32x4 g0;
  g0.x = 1u;                                        // count=1, user mode, no gather
  g0.y = lds_addr;                                  // LDS byte address
  g0.z = (unsigned)(gaddr & 0xFFFFFFFFull);         // global_addr[31:0]
  g0.w = (unsigned)((gaddr >> 32) & 0x01FFFFFFull)  // global_addr[56:32]
         | 0x80000000u;                             // type=2 ("image")
  i32x8 g1;
  g1[0] = (int)0x00010000u;                         // data_size=1 -> 2 bytes
  g1[1] = (int)((td0 & 0xFFFFu) << 16);             // tensor_dim0[15:0]
  g1[2] = (int)((td0 >> 16) | ((td1 & 0xFFFFu) << 16));
  g1[3] = (int)((td1 >> 16) | (tile0 << 16));       // tile_dim0
  g1[4] = (int)(tile1 & 0xFFFFu);                   // tile_dim1 (tile_dim2=0)
  g1[5] = (int)stride0;                             // tensor_dim0_stride[31:0]
  g1[6] = 0;
  g1[7] = 0;
  i32x4 z = {0, 0, 0, 0};
#if defined(__clang_major__) && (__clang_major__ >= 23)
  i32x8 z8 = {0, 0, 0, 0, 0, 0, 0, 0};
  __builtin_amdgcn_tensor_load_to_lds(g0, g1, z, z, z8, 0);
#else
  __builtin_amdgcn_tensor_load_to_lds(g0, g1, z, z, 0);
#endif
}
#endif // CDNA5_HAS_TDM

// Problem constants
#define BB 64
#define CC 256
#define SS 784           // 28*28
#define KK 128
#define EE 16
#define CKK 2304         // C*9
#define MIXK 384         // C + K

// d_out float offsets (tuple flattened in return order)
#define OFF_OUT    0
#define OFF_SCORES 12845056
#define OFF_IDXS   12846080
#define OFF_QF     12846208
#define OFF_EXP    12896384

// ---------------- 1) convert+pad x to bf16 [B][C][30][30] ----------------
__global__ void k_cvt_x(const float* __restrict__ x, unsigned short* __restrict__ xp, int total) {
  for (int i = blockIdx.x * blockDim.x + threadIdx.x; i < total; i += gridDim.x * blockDim.x) {
    int c900 = i % 900;
    int bc   = i / 900;
    int y = c900 / 30, xx = c900 % 30;
    float v = 0.f;
    if (y >= 1 && y <= 28 && xx >= 1 && xx <= 28)
      v = x[(size_t)bc * SS + (y - 1) * 28 + (xx - 1)];
    xp[i] = f2bf(v);
  }
}

// ---------------- 2) convert weights to bf16 ----------------
__global__ void k_cvt_w(const float* __restrict__ wsp_f, const float* __restrict__ wch_f,
                        unsigned short* __restrict__ wsp, unsigned short* __restrict__ wch,
                        int n_sp, int n_ch) {
  int stride = gridDim.x * blockDim.x;
  int t0 = blockIdx.x * blockDim.x + threadIdx.x;
  for (int i = t0; i < n_sp; i += stride) wsp[i] = f2bf(wsp_f[i]);
  for (int i = t0; i < n_ch; i += stride) wch[i] = f2bf(wch_f[i]);
}

// ---------------- 3) gate: mean -> norm -> dist -> softmax -> top2 ----------------
__global__ void __launch_bounds__(256)
k_gate(const float* __restrict__ x, const float* __restrict__ proto,
       float* __restrict__ out, int* __restrict__ sel_e, float* __restrict__ sel_s) {
  const int b = blockIdx.x, tid = threadIdx.x;
  __shared__ float q[SS];
  __shared__ float red[256];
  __shared__ float accum[32];

  for (int p = tid; p < SS; p += 256) {
    float s = 0.f;
    const float* xb = x + (size_t)b * CC * SS + p;
    for (int c = 0; c < CC; ++c) s += xb[c * SS];
    float qv = s * (1.0f / 256.0f);
    q[p] = qv;
    out[OFF_QF + b * SS + p] = qv;
  }
  __syncthreads();

  float local = 0.f;
  for (int p = tid; p < SS; p += 256) local += q[p] * q[p];
  red[tid] = local;
  __syncthreads();
  for (int s = 128; s > 0; s >>= 1) {
    if (tid < s) red[tid] += red[tid + s];
    __syncthreads();
  }
  float sumsq = red[0];
  __syncthreads();

  float d[EE], p2[EE];
#pragma unroll
  for (int e = 0; e < EE; ++e) { d[e] = 0.f; p2[e] = 0.f; }
  for (int p = tid; p < SS; p += 256) {
    float qv = q[p];
#pragma unroll
    for (int e = 0; e < EE; ++e) {
      float pr = proto[e * SS + p];
      d[e]  += qv * pr;
      p2[e] += pr * pr;
    }
  }
  if (tid < 32) accum[tid] = 0.f;
  __syncthreads();
#pragma unroll
  for (int e = 0; e < EE; ++e) {
    atomicAdd(&accum[e], d[e]);
    atomicAdd(&accum[16 + e], p2[e]);
  }
  __syncthreads();

  if (tid == 0) {
    float nrm = sqrtf(sumsq);
    float inv = 1.f / fmaxf(nrm, 1e-12f);
    float nq2 = sumsq * inv * inv;
    float dist[EE];
    float mx = -1e30f;
    for (int e = 0; e < EE; ++e) {
      float d2 = nq2 + accum[16 + e] - 2.f * inv * accum[e];
      dist[e] = sqrtf(fmaxf(d2, 0.f));
      mx = fmaxf(mx, dist[e]);
    }
    float se = 0.f;
    for (int e = 0; e < EE; ++e) { dist[e] = __expf(dist[e] - mx); se += dist[e]; }
    float b1 = -1e30f, b2 = -1e30f; int i1 = 0, i2 = 0;
    for (int e = 0; e < EE; ++e) {
      float sc = dist[e] / se;
      out[OFF_SCORES + b * EE + e] = sc;
      if (sc > b1)      { b2 = b1; i2 = i1; b1 = sc; i1 = e; }
      else if (sc > b2) { b2 = sc; i2 = e; }
    }
    out[OFF_IDXS + b * 2 + 0] = (float)i1;
    out[OFF_IDXS + b * 2 + 1] = (float)i2;
    sel_e[b * 2 + 0] = i1;  sel_e[b * 2 + 1] = i2;
    sel_s[b * 2 + 0] = b1;  sel_s[b * 2 + 1] = b2;
  }
}

// ---------------- 4) grouped 3x3 conv as implicit-im2col WMMA GEMM ----------------
// Per (b,t): M=128, N=784, Kred=2304. Block: 8 waves, tile M=128 x N=16.
// Double-buffered LDS, ONE barrier per k-step of 64; A tile via TDM DMA.
__global__ void __launch_bounds__(256)
k_conv(const unsigned short* __restrict__ xp, const unsigned short* __restrict__ wsp,
       const int* __restrict__ sel_e, unsigned short* __restrict__ nf) {
  const int n0 = blockIdx.x * 16;
  const int bt = blockIdx.y;
  const int b  = bt >> 1;
  const int tid = threadIdx.x;
  const int lane = tid & 31, wv = tid >> 5;
  const int e = sel_e[bt];

  __shared__ __align__(16) unsigned short sA[2][128 * 64]; // 2 x 16 KB, row-major [m][64]
  __shared__ __align__(16) unsigned short sB[2][16 * 64];  // 2 x 2 KB,  col-major [n][64]

#if !CDNA5_HAS_TDM
  const uint4* wsp4 = (const uint4*)(wsp + (size_t)e * KK * CKK);
#endif

  // ---- incremental im2col gather state: thread owns column n, kk in {kkb..kkb+3}
  const int n  = tid & 15;
  const int kkb = (tid >> 4) * 4;                   // 0..60
  int c  = kkb / 9;
  int r  = kkb - c * 9;
  int dy = r / 3;
  int dx = r - dy * 3;
  int off = c * 900 + dy * 30 + dx;                 // x_pad offset for k = kkb (one-time divs)
  const int p  = n0 + n;
  const int py = p / 28, px = p - py * 28;
  const unsigned short* xb = xp + (size_t)b * CC * 900 + py * 30 + px;

  auto gatherB = [&](int buf) {
    unsigned short v[4];
    int o = off, tdx = dx, tdy = dy;
#pragma unroll
    for (int j = 0; j < 4; ++j) {
      v[j] = xb[o];
      bool wx = (tdx == 2);
      bool wy = wx && (tdy == 2);
      o  += wy ? 838 : (wx ? 28 : 1);               // +1 within row, +28 row wrap, +838 chan wrap
      tdy = wx ? (wy ? 0 : tdy + 1) : tdy;
      tdx = wx ? 0 : tdx + 1;
      if (j == 0) { off = o + 6300; dx = tdx; dy = tdy; }  // base advance: +64 k == +1 then +7*900
    }
    unsigned long long pk = (unsigned long long)v[0]
                          | ((unsigned long long)v[1] << 16)
                          | ((unsigned long long)v[2] << 32)
                          | ((unsigned long long)v[3] << 48);
    *(unsigned long long*)&sB[buf][n * 64 + kkb] = pk;
    __builtin_prefetch((const void*)(xb + off), 0, 0);
  };

  auto loadA = [&](int buf, int k0) {
#if CDNA5_HAS_TDM
    if (wv == 0) {
      unsigned long long ga = (unsigned long long)(size_t)wsp
                            + ((size_t)e * KK * CKK + (size_t)k0) * 2ull;
      tdm_load_2d((unsigned)(size_t)&sA[buf][0], ga,
                  /*td0=*/64, /*td1=*/128, /*tile0=*/64, /*tile1=*/128,
                  /*stride0=*/CKK);
    }
#else
    int kq = k0 >> 3;
    uint4* sA4 = (uint4*)&sA[buf][0];
#pragma unroll
    for (int j = 0; j < 4; ++j) {
      int i = tid + j * 256;
      int m = i >> 3, cu = i & 7;
      sA4[i] = wsp4[m * 288 + kq + cu];
    }
#endif
  };

  v8f acc0 = {}, acc1 = {};

  // prologue: fill buffer 0
  loadA(0, 0);
  gatherB(0);
#if CDNA5_HAS_TDM
  if (wv == 0) __builtin_amdgcn_s_wait_tensorcnt(0);
#endif
  __syncthreads();

  for (int s = 0; s < 36; ++s) {
    const int buf = s & 1;
    if (s + 1 < 36) {
      loadA(buf ^ 1, (s + 1) * 64);
      gatherB(buf ^ 1);
    }
    const char* arow = (const char*)&sA[buf][0] + (wv * 16 + (lane & 15)) * 128 + ((lane >> 4) * 16);
    const char* bcol = (const char*)&sB[buf][0] + (lane & 15) * 128 + ((lane >> 4) * 16);
    {
      uint4 alo = *(const uint4*)(arow);
      uint4 ahi = *(const uint4*)(arow + 32);
      uint4 blo = *(const uint4*)(bcol);
      uint4 bhi = *(const uint4*)(bcol + 32);
      acc0 = __builtin_amdgcn_wmma_f32_16x16x32_bf16(
          false, make_frag(alo, ahi), false, make_frag(blo, bhi),
          (short)0, acc0, false, false);
    }
    {
      uint4 alo = *(const uint4*)(arow + 64);
      uint4 ahi = *(const uint4*)(arow + 96);
      uint4 blo = *(const uint4*)(bcol + 64);
      uint4 bhi = *(const uint4*)(bcol + 96);
      acc1 = __builtin_amdgcn_wmma_f32_16x16x32_bf16(
          false, make_frag(alo, ahi), false, make_frag(blo, bhi),
          (short)0, acc1, false, false);
    }
#if CDNA5_HAS_TDM
    if (wv == 0) __builtin_amdgcn_s_wait_tensorcnt(0);
#endif
    __syncthreads();
  }

  v8f acc = acc0 + acc1;
  unsigned short* nfb = nf + (size_t)bt * KK * SS;
  int nn = n0 + (lane & 15);
#pragma unroll
  for (int v = 0; v < 8; ++v) {
    int m = wv * 16 + (lane >> 4) * 8 + v;
    nfb[m * SS + nn] = f2bf(acc[v]);
  }
}

// ---------------- 5) 1x1 channel mix for both experts + weighted sum ----------------
// Per (b, n-tile): M=256, N=16, Kred=384, both topk experts in-block. k-step 32.
// Loop split at k=256: x-phase (8 steps, value shared across experts) then nf-phase (4 steps).
__global__ void __launch_bounds__(256)
k_mix(const unsigned short* __restrict__ xp, const unsigned short* __restrict__ wch,
      const unsigned short* __restrict__ nf, const int* __restrict__ sel_e,
      const float* __restrict__ sel_s, float* __restrict__ out) {
  const int n0 = blockIdx.x * 16;
  const int b  = blockIdx.y;
  const int tid = threadIdx.x, lane = tid & 31, wv = tid >> 5;
  const int e0 = sel_e[b * 2], e1 = sel_e[b * 2 + 1];
  const float s0 = sel_s[b * 2], s1 = sel_s[b * 2 + 1];

  __shared__ __align__(16) unsigned short sA[2 * 256 * 32]; // 32 KB: [t][m][32]
  __shared__ __align__(16) unsigned short sB[2 * 16 * 32];  // 2 KB:  [t][n][32] col-major

#if !CDNA5_HAS_TDM
  const uint4* wch4 = (const uint4*)wch;
#endif

  // gather state: thread owns column n for (t in {0,1}) x (kk in {kk0, kk0+16})
  const int n = tid & 15, kk0 = tid >> 4;
  const int p = n0 + n;
  const int py = p / 28, px = p - py * 28;
  const unsigned short* xk  = xp + (size_t)b * CC * 900 + (size_t)kk0 * 900 + (py + 1) * 30 + (px + 1);
  const unsigned short* nf0p = nf + ((size_t)(b * 2 + 0) * KK + kk0) * SS + p;
  const unsigned short* nf1p = nf + ((size_t)(b * 2 + 1) * KK + kk0) * SS + p;
  unsigned short* d0 = sB + n * 32 + kk0;        // t=0 slot
  unsigned short* d1 = sB + 512 + n * 32 + kk0;  // t=1 slot

  v8f acc[2][2] = {};

  for (int s = 0; s < 12; ++s) {
    const int k0 = s * 32;
#if CDNA5_HAS_TDM
    if (wv == 0) {
      unsigned long long ga0 = (unsigned long long)(size_t)wch
                             + ((size_t)e0 * CC * MIXK + (size_t)k0) * 2ull;
      unsigned long long ga1 = (unsigned long long)(size_t)wch
                             + ((size_t)e1 * CC * MIXK + (size_t)k0) * 2ull;
      tdm_load_2d((unsigned)(size_t)sA,          ga0, 32, 256, 32, 256, MIXK);
      tdm_load_2d((unsigned)(size_t)sA + 16384u, ga1, 32, 256, 32, 256, MIXK);
    }
#else
    {
      int kq = k0 >> 3;
      uint4* sA4 = (uint4*)sA;
#pragma unroll
      for (int j = 0; j < 8; ++j) {
        int i = tid + j * 256;
        int t = i >> 10;
        int rem = i & 1023;
        int m = rem >> 2, cu = rem & 3;
        int e = t ? e1 : e0;
        sA4[i] = wch4[((size_t)e * CC + m) * 48 + kq + cu];
      }
    }
#endif
    if (s < 8) {
      // rows k0..k0+31 come from x: identical for both experts
      unsigned short v0 = xk[0];
      unsigned short v1 = xk[16 * 900];
      d0[0]  = v0; d1[0]  = v0;
      d0[16] = v1; d1[16] = v1;
      xk += 32 * 900;
    } else {
      // rows come from nf, per expert
      unsigned short v00 = nf0p[0], v01 = nf0p[16 * SS];
      unsigned short v10 = nf1p[0], v11 = nf1p[16 * SS];
      d0[0] = v00; d0[16] = v01;
      d1[0] = v10; d1[16] = v11;
      nf0p += 32 * SS; nf1p += 32 * SS;
    }
#if CDNA5_HAS_TDM
    if (wv == 0) __builtin_amdgcn_s_wait_tensorcnt(0);
#endif
    __syncthreads();
#pragma unroll
    for (int t = 0; t < 2; ++t) {
      const char* bcol = (const char*)sB + t * 1024 + (lane & 15) * 64 + ((lane >> 4) * 16);
      uint4 blo = *(const uint4*)(bcol);
      uint4 bhi = *(const uint4*)(bcol + 32);
      v16bf bf = make_frag(blo, bhi);
#pragma unroll
      for (int mi = 0; mi < 2; ++mi) {
        int st = mi ? (wv + 8) : wv;
        const char* arow = (const char*)sA + t * 16384 + (st * 16 + (lane & 15)) * 64 + ((lane >> 4) * 16);
        uint4 alo = *(const uint4*)(arow);
        uint4 ahi = *(const uint4*)(arow + 32);
        acc[t][mi] = __builtin_amdgcn_wmma_f32_16x16x32_bf16(
            false, make_frag(alo, ahi), false, bf, (short)0, acc[t][mi], false, false);
      }
    }
    __syncthreads();
  }

  int nn = n0 + (lane & 15);
#pragma unroll
  for (int mi = 0; mi < 2; ++mi) {
    int st = mi ? (wv + 8) : wv;
#pragma unroll
    for (int v = 0; v < 8; ++v) {
      int m = st * 16 + (lane >> 4) * 8 + v;
      float a0 = acc[0][mi][v], a1 = acc[1][mi][v];
      out[OFF_EXP + (((size_t)b * 2 + 0) * CC + m) * SS + nn] = a0;
      out[OFF_EXP + (((size_t)b * 2 + 1) * CC + m) * SS + nn] = a1;
      out[OFF_OUT + ((size_t)b * CC + m) * SS + nn] = s0 * a0 + s1 * a1;
    }
  }
}

// ---------------- launch ----------------
extern "C" void kernel_launch(void* const* d_in, const int* in_sizes, int n_in,
                              void* d_out, int out_size, void* d_ws, size_t ws_size,
                              hipStream_t stream) {
  (void)in_sizes; (void)n_in; (void)out_size; (void)ws_size;
  // inputs: 0=pretrained_x (unused), 1=x, 2=conv_special_param, 3=conv_channel_param,
  //         4=prototype, 5=topk (==2, hardcoded)
  const float* x      = (const float*)d_in[1];
  const float* wsp_f  = (const float*)d_in[2];
  const float* wch_f  = (const float*)d_in[3];
  const float* proto  = (const float*)d_in[4];
  float* out = (float*)d_out;

  char* ws = (char*)d_ws;
  unsigned short* xp  = (unsigned short*)(ws + 0);          // 64*256*900*2  = 29,491,200
  unsigned short* wsp = (unsigned short*)(ws + 29491200);   // 16*128*2304*2 =  9,437,184
  unsigned short* wch = (unsigned short*)(ws + 38928384);   // 16*256*384*2  =  3,145,728
  unsigned short* nf  = (unsigned short*)(ws + 42074112);   // 64*2*128*784*2 = 25,690,112
  int*   sel_e = (int*)(ws + 67764224);                     // 128 ints
  float* sel_s = (float*)(ws + 67764736);                   // 128 floats

  k_cvt_x<<<dim3(2048), dim3(256), 0, stream>>>(x, xp, BB * CC * 900);
  k_cvt_w<<<dim3(2048), dim3(256), 0, stream>>>(wsp_f, wch_f, wsp, wch,
                                                EE * KK * CKK, EE * CC * MIXK);
  k_gate<<<dim3(BB), dim3(256), 0, stream>>>(x, proto, out, sel_e, sel_s);
  k_conv<<<dim3(49, BB * 2), dim3(256), 0, stream>>>(xp, wsp, sel_e, nf);
  k_mix<<<dim3(49, BB), dim3(256), 0, stream>>>(xp, wch, nf, sel_e, sel_s, out);
}
